// WHNNAggregator_43258910605642
// MI455X (gfx1250) — matrix operand order repaired and checked by reading, not compiled, not run on hardware
//
#include <hip/hip_runtime.h>
#include <hip/hip_bf16.h>

// ---------------------------------------------------------------------------
// WHNN aggregator for MI455X (gfx1250), bs=2, C=256, N=4096, E=2048, P=32.
// Heavy contractions use v_wmma_f32_16x16x32_bf16 with double-buffered LDS.
// ---------------------------------------------------------------------------

#define BS 2
#define CC 256
#define NN 4096
#define EE 2048
#define PP 32

typedef __bf16 v16bf __attribute__((ext_vector_type(16)));
typedef __bf16 v8bf  __attribute__((ext_vector_type(8)));
typedef __bf16 v4bf  __attribute__((ext_vector_type(4)));
typedef float  v8f   __attribute__((ext_vector_type(8)));

__device__ __forceinline__ __bf16 f2bf(float f) {
    return (__bf16)f;   // hardware v_cvt (bf16 is a native type on gfx1250)
}

// A fragment (16x32 bf16): lane<16 holds M=l16, K={0..7,16..23}; lane>=16 K={8..15,24..31}
__device__ __forceinline__ v16bf load_frag_a(const __bf16* base, int row, int hi) {
    union { v16bf v; v8bf h[2]; } u;
    const char* p = (const char*)base + row * 64 + hi * 16;
    u.h[0] = *(const v8bf*)(p);
    u.h[1] = *(const v8bf*)(p + 32);
    return u.v;
}
// B fragment (32x16 bf16, staged transposed as [n][k]): lane<16 holds N=l16, K={0..7,16..23};
// lane>=16 holds N=l16, K={8..15,24..31}.
__device__ __forceinline__ v16bf load_frag_b(const __bf16* base, int row, int hi) {
    union { v16bf v; v8bf h[2]; } u;
    const char* p = (const char*)base + row * 64 + hi * 16;
    u.h[0] = *(const v8bf*)(p);
    u.h[1] = *(const v8bf*)(p + 32);
    return u.v;
}

// ---------------------------------------------------------------------------
// Generic tiled GEMM  D[m,n] = sum_k A[m,k]*B[k,n]   (f32 in, bf16 WMMA, f32 out)
//   A_ROW: A row-major (A[m*lda+k]) else col-major (A[k*lda+m])
//   B_ROW: B row-major (B[k*ldb+n]) else col-major (B[n*ldb+k])
//   EPI 0: D[m*ldd+n]=v
//   EPI 1: D[m*ldd+n] = -(auxM[m]+auxN[n]-2v)/(tau+1e-8)   (fused cost->logit)
//   EPI 2: D[n*ldd+m]=v (transposed store)
// Tile: BM=BN=64, BK=32, 128 threads = 4 waves, each wave a 32x32 region.
// Double-buffered LDS: one barrier per K-step, staging overlaps WMMA.
// ---------------------------------------------------------------------------
template <bool A_ROW, bool B_ROW, int EPI>
__global__ __launch_bounds__(128)
void gemm_bf16_wmma(const float* __restrict__ A, const float* __restrict__ B,
                    float* __restrict__ D,
                    const float* __restrict__ auxM, const float* __restrict__ auxN,
                    int M, int Ncol, int Kdim, int lda, int ldb, int ldd,
                    long Abs, long Bbs, long Dbs, int auxMbs, int auxNbs)
{
    constexpr int BM = 64, BN = 64, BK = 32;
    __shared__ __align__(16) __bf16 Al[2][BM * BK];   // [m][k]
    __shared__ __align__(16) __bf16 Bl[2][BN * BK];   // [n][k] (transposed stage)

    const int tid = threadIdx.x;
    const int b   = blockIdx.z;
    const int m0  = blockIdx.x * BM;
    const int n0  = blockIdx.y * BN;
    A += (long)b * Abs;  B += (long)b * Bbs;  D += (long)b * Dbs;

    const int wave = tid >> 5, lane = tid & 31;
    const int wm = (wave >> 1) * 32, wn = (wave & 1) * 32;
    const int l16 = lane & 15, hi = lane >> 4;

    // ---- tile staging (global f32 -> LDS bf16) -------------------------------
    auto stageA = [&](int k0, __bf16* Ab) {
        if (A_ROW) {
            #pragma unroll
            for (int i = 0; i < 4; ++i) {               // 512 float4s, k-contiguous
                int linear = i * 128 + tid;
                int kq = linear & 7, m = linear >> 3;
                float4 v = *(const float4*)(A + (long)(m0 + m) * lda + k0 + kq * 4);
                v4bf t; t[0] = f2bf(v.x); t[1] = f2bf(v.y); t[2] = f2bf(v.z); t[3] = f2bf(v.w);
                *(v4bf*)&Ab[m * 32 + kq * 4] = t;
            }
        } else {
            #pragma unroll
            for (int i = 0; i < 4; ++i) {               // 512 float4s, m-contiguous
                int linear = i * 128 + tid;
                int mq = linear & 15, k = linear >> 4;
                int m = mq * 4;
                float4 v = *(const float4*)(A + (long)(k0 + k) * lda + m0 + m);
                Ab[(m + 0) * 32 + k] = f2bf(v.x);
                Ab[(m + 1) * 32 + k] = f2bf(v.y);
                Ab[(m + 2) * 32 + k] = f2bf(v.z);
                Ab[(m + 3) * 32 + k] = f2bf(v.w);
            }
        }
    };
    auto stageB = [&](int k0, __bf16* Bb) {
        if (B_ROW) {
            #pragma unroll
            for (int i = 0; i < 4; ++i) {               // 512 float4s, j-contiguous
                int linear = i * 128 + tid;
                int jq = linear & 15, k = linear >> 4;
                int j = jq * 4;
                int jg = n0 + j;
                float4 v;
                if (jg < Ncol) {                        // Ncol is a multiple of 4
                    v = *(const float4*)(B + (long)(k0 + k) * ldb + jg);
                } else {
                    v.x = v.y = v.z = v.w = 0.0f;
                }
                Bb[(j + 0) * 32 + k] = f2bf(v.x);
                Bb[(j + 1) * 32 + k] = f2bf(v.y);
                Bb[(j + 2) * 32 + k] = f2bf(v.z);
                Bb[(j + 3) * 32 + k] = f2bf(v.w);
            }
        } else {
            #pragma unroll
            for (int i = 0; i < 4; ++i) {               // 512 float4s, k-contiguous
                int linear = i * 128 + tid;
                int kq = linear & 7, n = linear >> 3;
                float4 v = *(const float4*)(B + (long)(n0 + n) * ldb + k0 + kq * 4);
                v4bf t; t[0] = f2bf(v.x); t[1] = f2bf(v.y); t[2] = f2bf(v.z); t[3] = f2bf(v.w);
                *(v4bf*)&Bb[n * 32 + kq * 4] = t;
            }
        }
    };

    v8f acc[2][2] = {};

    stageA(0, Al[0]);
    stageB(0, Bl[0]);
    int cur = 0;

    for (int k0 = 0; k0 < Kdim; k0 += BK) {
        __syncthreads();                                 // staged buffer `cur` ready

        v16bf a0 = load_frag_a(Al[cur], wm + l16, hi);
        v16bf a1 = load_frag_a(Al[cur], wm + 16 + l16, hi);
        v16bf b0 = load_frag_b(Bl[cur], wn + l16, hi);
        v16bf b1 = load_frag_b(Bl[cur], wn + 16 + l16, hi);

        const int kn = k0 + BK;
        if (kn < Kdim) {                                 // stage next tile into other buffer
            const float* pa = A_ROW ? (A + (long)m0 * lda + kn + BK)
                                    : (A + (long)(kn + BK) * lda + m0);
            const float* pb = B_ROW ? (B + (long)(kn + BK) * ldb + n0)
                                    : (B + (long)n0 * ldb + kn + BK);
            __builtin_prefetch(pa, 0, 1);                // global_prefetch_b8, 2 steps ahead
            __builtin_prefetch(pb, 0, 1);
            stageA(kn, Al[cur ^ 1]);
            stageB(kn, Bl[cur ^ 1]);
        }

        acc[0][0] = __builtin_amdgcn_wmma_f32_16x16x32_bf16(false, a0, false, b0, (short)0, acc[0][0], false, false);
        acc[0][1] = __builtin_amdgcn_wmma_f32_16x16x32_bf16(false, a0, false, b1, (short)0, acc[0][1], false, false);
        acc[1][0] = __builtin_amdgcn_wmma_f32_16x16x32_bf16(false, a1, false, b0, (short)0, acc[1][0], false, false);
        acc[1][1] = __builtin_amdgcn_wmma_f32_16x16x32_bf16(false, a1, false, b1, (short)0, acc[1][1], false, false);

        cur ^= 1;
    }

    const float kNegInvTau = -1.0f / (0.1f + 1e-8f);
    #pragma unroll
    for (int mi = 0; mi < 2; ++mi) {
        #pragma unroll
        for (int ni = 0; ni < 2; ++ni) {
            int ng = n0 + wn + ni * 16 + l16;
            if (ng < Ncol) {
                #pragma unroll
                for (int r = 0; r < 8; ++r) {
                    int mg = m0 + wm + mi * 16 + r + hi * 8;
                    float v = acc[mi][ni][r];
                    if (EPI == 0) {
                        D[(long)mg * ldd + ng] = v;
                    } else if (EPI == 1) {
                        float c = auxM[(long)auxMbs * b + mg] + auxN[(long)auxNbs * b + ng] - 2.0f * v;
                        D[(long)mg * ldd + ng] = c * kNegInvTau;
                    } else {
                        D[(long)ng * ldd + mg] = v;
                    }
                }
            }
        }
    }
}

// ---------------------------------------------------------------------------
// proj (1x1 conv) + eval BatchNorm + per-vertex squared norm.
// One thread per (b,n); proj_w cached transposed in LDS ([c][p]).
// ---------------------------------------------------------------------------
__global__ __launch_bounds__(256)
void proj_bn_kernel(const float* __restrict__ vf, const float* __restrict__ w,
                    const float* __restrict__ gamma, const float* __restrict__ beta,
                    const float* __restrict__ mean, const float* __restrict__ var,
                    float* __restrict__ K, float* __restrict__ Ksq)
{
    __shared__ __align__(16) float wl[CC * PP];  // 32 KB
    __shared__ float sc[PP], bi[PP];
    const int tid = threadIdx.x;
    #pragma unroll
    for (int i = 0; i < 32; ++i) {
        int idx = tid + i * 256;               // idx = p*256 + c (coalesced read)
        int p = idx >> 8, c = idx & 255;
        wl[c * 32 + p] = w[idx];
    }
    if (tid < PP) {
        float s = gamma[tid] * rsqrtf(var[tid] + 1e-5f);
        sc[tid] = s;
        bi[tid] = beta[tid] - mean[tid] * s;
    }
    __syncthreads();

    const int flat = blockIdx.x * 256 + tid;   // over bs*N
    const int b = flat >> 12;                  // N = 4096
    const int n = flat & (NN - 1);
    const float* v = vf + (long)b * CC * NN + n;

    float acc[PP];
    #pragma unroll
    for (int p = 0; p < PP; ++p) acc[p] = 0.0f;

    for (int c = 0; c < CC; ++c) {
        float x = v[(long)c * NN];
        const float4* wr = (const float4*)&wl[c * 32];
        #pragma unroll
        for (int q = 0; q < 8; ++q) {
            float4 ww = wr[q];
            acc[4 * q + 0] += ww.x * x;
            acc[4 * q + 1] += ww.y * x;
            acc[4 * q + 2] += ww.z * x;
            acc[4 * q + 3] += ww.w * x;
        }
    }

    float ksum = 0.0f;
    #pragma unroll
    for (int p = 0; p < PP; ++p) {
        float kv = acc[p] * sc[p] + bi[p];
        acc[p] = kv;
        ksum += kv * kv;
    }
    float* kout = K + (long)flat * PP;
    #pragma unroll
    for (int q = 0; q < 8; ++q) {
        float4 o; o.x = acc[4 * q]; o.y = acc[4 * q + 1]; o.z = acc[4 * q + 2]; o.w = acc[4 * q + 3];
        *(float4*)(kout + 4 * q) = o;
    }
    Ksq[flat] = ksum;
}

// per-row squared norm of [rows, 32] matrix
__global__ __launch_bounds__(256)
void rowsq32_kernel(const float* __restrict__ X, float* __restrict__ out, int rows)
{
    int i = blockIdx.x * 256 + threadIdx.x;
    if (i < rows) {
        const float4* p = (const float4*)(X + (long)i * 32);
        float s = 0.0f;
        #pragma unroll
        for (int q = 0; q < 8; ++q) {
            float4 v = p[q];
            s += v.x * v.x + v.y * v.y + v.z * v.z + v.w * v.w;
        }
        out[i] = s;
    }
}

// in-place softmax along rows of length Nlen (one block per row)
__global__ __launch_bounds__(256)
void softmax_rows_kernel(float* __restrict__ S, int Nlen)
{
    float* row = S + (long)blockIdx.x * Nlen;
    __shared__ float red[256];
    const int tid = threadIdx.x;

    float mx = -3.4e38f;
    for (int i = tid; i < Nlen; i += 256) mx = fmaxf(mx, row[i]);
    red[tid] = mx; __syncthreads();
    for (int s = 128; s > 0; s >>= 1) {
        if (tid < s) red[tid] = fmaxf(red[tid], red[tid + s]);
        __syncthreads();
    }
    mx = red[0]; __syncthreads();

    float sum = 0.0f;
    for (int i = tid; i < Nlen; i += 256) {
        float e = __expf(row[i] - mx);
        row[i] = e;
        sum += e;
    }
    red[tid] = sum; __syncthreads();
    for (int s = 128; s > 0; s >>= 1) {
        if (tid < s) red[tid] += red[tid + s];
        __syncthreads();
    }
    float inv = 1.0f / red[0];
    for (int i = tid; i < Nlen; i += 256) row[i] *= inv;
}

// ---------------------------------------------------------------------------
extern "C" void kernel_launch(void* const* d_in, const int* in_sizes, int n_in,
                              void* d_out, int out_size, void* d_ws, size_t ws_size,
                              hipStream_t stream)
{
    (void)in_sizes; (void)n_in; (void)out_size; (void)ws_size;
    const float* vf    = (const float*)d_in[0];   // [bs,C,N]
    const float* inc   = (const float*)d_in[1];   // [bs,N,E]
    const float* ied   = (const float*)d_in[2];   // [bs,E,E]
    const float* pw    = (const float*)d_in[5];   // [P,C]
    const float* gam   = (const float*)d_in[6];
    const float* bet   = (const float*)d_in[7];
    const float* mea   = (const float*)d_in[8];
    const float* var   = (const float*)d_in[9];
    float* out = (float*)d_out;                   // [bs,C,E]

    float* ws   = (float*)d_ws;
    float* K    = ws;                                   // bs*N*P     = 262144
    float* Ksq  = K   + (long)BS * NN * PP;             // bs*N       = 8192
    float* P1   = Ksq + (long)BS * NN;                  // bs*E*P     = 131072
    float* PR   = P1  + (long)BS * EE * PP;             // bs*E*P     = 131072
    float* PSQ  = PR  + (long)BS * EE * PP;             // bs*E       = 4096
    float* S    = PSQ + (long)BS * EE;                  // bs*E*N     = 16777216
    float* EF   = S   + (long)BS * EE * NN;             // bs*E*C     = 1048576

    // 1) K = BN(proj_w @ vf)   -> K[b,n,p], Ksq[b,n]
    proj_bn_kernel<<<dim3((BS * NN) / 256), 256, 0, stream>>>(vf, pw, gam, bet, mea, var, K, Ksq);

    // 2) proto1[e,p] = sum_n inc[n,e] * K[n,p]    (A col-major, B row-major)
    gemm_bf16_wmma<false, true, 0><<<dim3(EE / 64, 1, BS), 128, 0, stream>>>(
        inc, K, P1, nullptr, nullptr,
        EE, PP, NN, /*lda*/EE, /*ldb*/PP, /*ldd*/PP,
        (long)NN * EE, (long)NN * PP, (long)EE * PP, 0, 0);

    // 3) proto[e,p] = sum_f ied[e,f] * proto1[f,p]
    gemm_bf16_wmma<true, true, 0><<<dim3(EE / 64, 1, BS), 128, 0, stream>>>(
        ied, P1, PR, nullptr, nullptr,
        EE, PP, EE, EE, PP, PP,
        (long)EE * EE, (long)EE * PP, (long)EE * PP, 0, 0);

    // 4) proto squared norms
    rowsq32_kernel<<<dim3((BS * EE) / 256), 256, 0, stream>>>(PR, PSQ, BS * EE);

    // 5) logits S[e,n] = -(psq[e] + ksq[n] - 2 * proto.K^T) / tau   (B = K col-major)
    gemm_bf16_wmma<true, false, 1><<<dim3(EE / 64, NN / 64, BS), 128, 0, stream>>>(
        PR, K, S, PSQ, Ksq,
        EE, NN, PP, PP, PP, NN,
        (long)EE * PP, (long)NN * PP, (long)EE * NN, EE, NN);

    // 6) softmax over N per (b,e) row -> weights (in place)
    softmax_rows_kernel<<<dim3(BS * EE), 256, 0, stream>>>(S, NN);

    // 7) edge_feat[e,c] = sum_n w[e,n] * vf[c,n]   (B = vf col-major)
    gemm_bf16_wmma<true, false, 0><<<dim3(EE / 64, CC / 64, BS), 128, 0, stream>>>(
        S, vf, EF, nullptr, nullptr,
        EE, CC, NN, NN, NN, CC,
        (long)EE * NN, (long)CC * NN, (long)EE * CC, 0, 0);

    // 8) out[c,e] = sum_f ied[e,f] * ef[f,c]   (transposed store straight to d_out)
    gemm_bf16_wmma<true, true, 2><<<dim3(EE / 64, CC / 64, BS), 128, 0, stream>>>(
        ied, EF, out, nullptr, nullptr,
        EE, CC, EE, EE, CC, EE,
        (long)EE * EE, (long)EE * CC, (long)CC * EE, 0, 0);
}
